// MultiHeadAttention_46411416600738
// MI455X (gfx1250) — matrix operand order repaired and checked by reading, not compiled, not run on hardware
//
#include <hip/hip_runtime.h>

typedef __attribute__((ext_vector_type(16))) _Float16 v16h;
typedef __attribute__((ext_vector_type(8)))  _Float16 v8h;
typedef __attribute__((ext_vector_type(4)))  _Float16 v4h;
typedef __attribute__((ext_vector_type(2)))  _Float16 v2h;
typedef __attribute__((ext_vector_type(8)))  float    v8f;
typedef __attribute__((ext_vector_type(4)))  float    v4f;
typedef __attribute__((ext_vector_type(2)))  float    v2f;
typedef __attribute__((ext_vector_type(4)))  int      v4i;
typedef __attribute__((ext_vector_type(8)))  int      v8i;
typedef __attribute__((ext_vector_type(4)))  unsigned int v4u;

constexpr int CB = 4;      // batch
constexpr int CS = 2048;   // seq len (q and k)
constexpr int CE = 1024;   // embed
constexpr int CH = 16;     // heads
constexpr int CD = 64;     // head dim

// --- CDNA5 async global->LDS (ASYNCcnt), guarded -----------------------------
#if defined(__HIP_DEVICE_COMPILE__) && \
    __has_builtin(__builtin_amdgcn_global_load_async_to_lds_b128) && \
    __has_builtin(__builtin_amdgcn_s_wait_asynccnt)
#define HAVE_ASYNC_LDS 1
typedef v4i __attribute__((address_space(1)))* gptr_v4i;   // global int4*
typedef v4i __attribute__((address_space(3)))* lptr_v4i;   // LDS int4*
#else
#define HAVE_ASYNC_LDS 0
#endif

// --- CDNA5 Tensor Data Mover (TENSORcnt), guarded ----------------------------
#if defined(__HIP_DEVICE_COMPILE__) && \
    __has_builtin(__builtin_amdgcn_tensor_load_to_lds) && \
    __has_builtin(__builtin_amdgcn_s_wait_tensorcnt)
#define HAVE_TDM 1
#else
#define HAVE_TDM 0
#endif

// ---------------------------------------------------------------------------
// Kernel 1: fused Q/K/V projection.  y[b,s,f] = sum_e x[b,s,e] * W[f,e] + b[f]
// A = X (M=B*S, K=E) converted f32->f16 into LDS; B = W^T staged k-major so
// the B fragment (lane=N, elements=K) is a strided gather down one Bs column.
// WG 256 = 8 waves; tile 128x128; wave grid 4(M) x 2(N) -> 2x4 accumulators.
// ---------------------------------------------------------------------------
__global__ __launch_bounds__(256) void proj_qkv_kernel(
    const float* __restrict__ query, const float* __restrict__ key,
    const float* __restrict__ value,
    const float* __restrict__ Wq, const float* __restrict__ Wk,
    const float* __restrict__ Wv,
    const float* __restrict__ bq, const float* __restrict__ bk,
    const float* __restrict__ bv,
    _Float16* __restrict__ q16, _Float16* __restrict__ k16,
    _Float16* __restrict__ v16)
{
  __shared__ __align__(32) _Float16 As[128][40];   // A tile, converted f16
  __shared__ __align__(32) _Float16 Bs[32][144];   // W^T tile: Bs[k][n]

  const float* X; const float* W; const float* bias; _Float16* dst;
  if (blockIdx.z == 0)      { X = query; W = Wq; bias = bq; dst = q16; }
  else if (blockIdx.z == 1) { X = key;   W = Wk; bias = bk; dst = k16; }
  else                      { X = value; W = Wv; bias = bv; dst = v16; }

  const int tid  = threadIdx.x;
  const int lane = tid & 31, wid = tid >> 5;
  const int laneLo = lane & 15, laneHi = lane >> 4;
  const int wm = wid & 3, wn = wid >> 2;
  const int m0 = blockIdx.y * 128, n0 = blockIdx.x * 128;

  v8f acc[2][4] = {};

  for (int k0 = 0; k0 < CE; k0 += 32) {
    __syncthreads();
    // stage A: 128 rows x 32 cols, f32 -> f16, one ds_store_b64 per chunk
    #pragma unroll
    for (int p = 0; p < 4; ++p) {
      int l = p * 256 + tid;                 // 0..1023
      int row = l >> 3, ch = (l & 7) << 2;
      v4f x = *(const v4f*)(X + (size_t)(m0 + row) * CE + k0 + ch);
      v4h hx;
      #pragma unroll
      for (int i = 0; i < 4; ++i) hx[i] = (_Float16)x[i];
      *(v4h*)&As[row][ch] = hx;
    }
    // stage W^T: Bs[k][n] = W[n0+n][k0+k]
    #pragma unroll
    for (int p = 0; p < 4; ++p) {
      int l = p * 256 + tid;
      int n = l >> 3, ch = (l & 7) << 2;
      v4f w = *(const v4f*)(W + (size_t)(n0 + n) * CE + k0 + ch);
      #pragma unroll
      for (int i = 0; i < 4; ++i) Bs[ch + i][n] = (_Float16)w[i];
    }
    __syncthreads();

    v16h a[2], b[4];
    #pragma unroll
    for (int fm = 0; fm < 2; ++fm) {
      const _Float16* pa = &As[wm * 32 + fm * 16 + laneLo][8 * laneHi];
      #pragma unroll
      for (int hh = 0; hh < 8; ++hh) { a[fm][hh] = pa[hh]; a[fm][hh + 8] = pa[hh + 16]; }
    }
    #pragma unroll
    for (int fn = 0; fn < 4; ++fn)
      for (int hh = 0; hh < 8; ++hh) { b[fn][hh] = Bs[8 * laneHi + hh][wn * 64 + fn * 16 + laneLo]; b[fn][hh + 8] = Bs[16 + 8 * laneHi + hh][wn * 64 + fn * 16 + laneLo]; }
    #pragma unroll
    for (int fm = 0; fm < 2; ++fm)
      #pragma unroll
      for (int fn = 0; fn < 4; ++fn)
        acc[fm][fn] = __builtin_amdgcn_wmma_f32_16x16x32_f16(
            false, a[fm], false, b[fn], (short)0, acc[fm][fn], false, false);
  }

  // epilogue: bias + convert + scatter to head-major f16
  #pragma unroll
  for (int fm = 0; fm < 2; ++fm) {
    #pragma unroll
    for (int fn = 0; fn < 4; ++fn) {
      int n = n0 + wn * 64 + fn * 16 + laneLo;
      float bb = bias[n];
      int hh = n >> 6, dd = n & 63;
      #pragma unroll
      for (int r = 0; r < 8; ++r) {
        int m = m0 + wm * 32 + fm * 16 + r + 8 * laneHi;
        int bidx = m >> 11, ss = m & 2047;
        dst[(((size_t)bidx * CH + hh) * CS + ss) * CD + dd] =
            (_Float16)(acc[fm][fn][r] + bb);
      }
    }
  }
}

// ---------------------------------------------------------------------------
// Kernel 2: attention for one (b, h, 32-row q tile).
//  - S^T = K * Q^T: A frags are row-major K read directly from global; Q^T
//    B frags staged once (via async global->LDS + LDS transpose) and kept in
//    VGPRs for the whole score loop (no barriers inside it).
//  - 32 x 2048 f32 score tile in LDS (257KB; CDNA5: 320KB/WGP).
//  - Softmax pass 3 writes normalized weights to HBM (b64 stores) and packs
//    them f16 in-place over the LDS row, so the P@V A-frags are plain
//    ds_load_b128 with zero converts in the 64-iteration loop.
// ---------------------------------------------------------------------------
__global__ __launch_bounds__(256) void attention_kernel(
    const _Float16* __restrict__ q16, const _Float16* __restrict__ k16,
    const _Float16* __restrict__ v16, const int* __restrict__ mask,
    float* __restrict__ attnW, _Float16* __restrict__ o16)
{
  __shared__ float Sc[32][2052];                  // 257KB score tile
  __shared__ __align__(32) _Float16 Qt[64][48];   // Q^T: Qt[d][q], 96B rows
  __shared__ __align__(16) _Float16 Qraw[32][64]; // async landing buffer, 4KB

  const int tid = threadIdx.x, lane = tid & 31, wid = tid >> 5;
  const int laneLo = lane & 15, laneHi = lane >> 4;
  const int q0 = blockIdx.x * 32;
  const int hd = blockIdx.y;
  const int bb = blockIdx.z;
  const size_t bh = (size_t)bb * CH + hd;
  const _Float16* Qg = q16 + (bh * CS + q0) * CD;
  const _Float16* Kg = k16 + bh * CS * CD;
  const _Float16* Vg = v16 + bh * CS * CD;

  // stage Q tile: async DMA to LDS (ASYNCcnt), then transpose LDS->LDS
  {
    int row = tid >> 3, ch = (tid & 7) << 3;     // row = q, ch = d0
#if HAVE_ASYNC_LDS
    __builtin_amdgcn_global_load_async_to_lds_b128(
        (gptr_v4i)(Qg + (size_t)row * CD + ch), (lptr_v4i)&Qraw[row][ch], 0, 0);
    __builtin_amdgcn_s_wait_asynccnt(0);
#else
    *(v8h*)&Qraw[row][ch] = *(const v8h*)(Qg + (size_t)row * CD + ch);
#endif
    __syncthreads();
    v8h v = *(const v8h*)&Qraw[row][ch];
    #pragma unroll
    for (int i = 0; i < 8; ++i) Qt[ch + i][row] = v[i];
  }
  __syncthreads();

  const float inv_scale = 0.125f;   // 1/sqrt(64)

  // ---- scores: S^T[kcol, q] = sum_d K[kcol,d] * Q[q,d] ----
  {
    v16h bq2[2][2];                 // 4 loop-invariant B fragments (Q^T)
    #pragma unroll
    for (int ks = 0; ks < 2; ++ks)
      #pragma unroll
      for (int nq = 0; nq < 2; ++nq)
        for (int hh = 0; hh < 8; ++hh) { bq2[ks][nq][hh] = Qt[ks * 32 + 8 * laneHi + hh][nq * 16 + laneLo]; bq2[ks][nq][hh + 8] = Qt[ks * 32 + 16 + 8 * laneHi + hh][nq * 16 + laneLo]; }

    for (int it = 0; it < 16; ++it) {
      int kb = it * 128 + wid * 16;              // this wave's 16 k-columns
      const _Float16* arow = Kg + (size_t)(kb + laneLo) * CD + 8 * laneHi;
      if (it + 1 < 16)                           // prefetch next K chunk
        __builtin_prefetch(arow + 128 * CD, 0, 0);
      v8f acc[2] = {};
      #pragma unroll
      for (int ks = 0; ks < 2; ++ks) {
        v8h lo = *(const v8h*)(arow + ks * 32);
        v8h hi = *(const v8h*)(arow + ks * 32 + 16);
        v16h a;
        #pragma unroll
        for (int hh = 0; hh < 8; ++hh) { a[hh] = lo[hh]; a[hh + 8] = hi[hh]; }
        #pragma unroll
        for (int nq = 0; nq < 2; ++nq)
          acc[nq] = __builtin_amdgcn_wmma_f32_16x16x32_f16(
              false, a, false, bq2[ks][nq], (short)0, acc[nq], false, false);
      }
      // C layout: m(kcol) = r + 8*laneHi, n(q) = laneLo -> transposed store
      #pragma unroll
      for (int nq = 0; nq < 2; ++nq)
        #pragma unroll
        for (int r = 0; r < 8; ++r)
          Sc[nq * 16 + laneLo][kb + r + 8 * laneHi] = acc[nq][r] * inv_scale;
    }
  }
  __syncthreads();

  // ---- masked stable softmax; weights -> HBM once; pack f16 in-place ----
  {
    const int* mbase = mask + ((size_t)bb * CS + q0) * CS;
    #pragma unroll
    for (int rr = 0; rr < 4; ++rr) {
      int row = wid * 4 + rr;                 // 8 waves x 4 rows = 32
      const int* mrow = mbase + (size_t)row * CS;
      float* srow = Sc[row];
      float mx = -3.0e38f;
      for (int c = lane; c < CS; c += 32) {
        float s = srow[c];
        if (mrow[c] == 0) s = -1.0e9f;
        srow[c] = s;
        mx = fmaxf(mx, s);
      }
      #pragma unroll
      for (int o = 16; o > 0; o >>= 1) mx = fmaxf(mx, __shfl_xor(mx, o, 32));
      float sum = 0.0f;
      for (int c = lane; c < CS; c += 32) {
        float e = __expf(srow[c] - mx);
        srow[c] = e;
        sum += e;
      }
      #pragma unroll
      for (int o = 16; o > 0; o >>= 1) sum += __shfl_xor(sum, o, 32);
      float inv = 1.0f / sum;
      float* wrow = attnW + (bh * CS + q0 + row) * CS;
      // each lane owns an adjacent column pair; packed write index c2 is
      // always below every future read index -> in-place pack is race-free
      for (int c2 = lane; c2 < CS / 2; c2 += 32) {
        float w0 = srow[2 * c2]     * inv;
        float w1 = srow[2 * c2 + 1] * inv;
        v2f wv = {w0, w1};
        *(v2f*)(wrow + 2 * c2) = wv;          // global_store_b64
        v2h pv = {(_Float16)w0, (_Float16)w1};
        ((v2h*)srow)[c2] = pv;                // packed f16, in-place
      }
    }
  }
  __syncthreads();

  // ---- out_head = P @ V ; 8 waves = 2(M frags) x 4(N frags), K = 2048 ----
  {
    const int fm = wid & 1, fn = wid >> 1;    // fn 0..3
    const _Float16* prow = (const _Float16*)Sc[fm * 16 + laneLo];
    v8f acc = {};
    for (int kk = 0; kk < CS; kk += 32) {
      const _Float16* pa = prow + kk + 8 * laneHi;
      v8h lo = *(const v8h*)pa;
      v8h hi = *(const v8h*)(pa + 16);
      v16h a;
      #pragma unroll
      for (int hh = 0; hh < 8; ++hh) { a[hh] = lo[hh]; a[hh + 8] = hi[hh]; }
      if (kk + 32 < CS)                        // prefetch next V rows
        __builtin_prefetch(Vg + (size_t)(kk + 32 + lane) * CD, 0, 0);
      v16h b;
      for (int hh = 0; hh < 8; ++hh) { b[hh] = Vg[(size_t)(kk + 8 * laneHi + hh) * CD + fn * 16 + laneLo]; b[hh + 8] = Vg[(size_t)(kk + 16 + 8 * laneHi + hh) * CD + fn * 16 + laneLo]; }
      acc = __builtin_amdgcn_wmma_f32_16x16x32_f16(
          false, a, false, b, (short)0, acc, false, false);
    }
    int col = hd * CD + fn * 16 + laneLo;     // back to (B,S,E) layout, f16
    #pragma unroll
    for (int r = 0; r < 8; ++r) {
      int q = q0 + fm * 16 + r + 8 * laneHi;
      o16[((size_t)bb * CS + q) * CE + col] = (_Float16)acc[r];
    }
  }
}

// ---------------------------------------------------------------------------
// Kernel 3: output projection.  out[m,n] = sum_k o16[m,k] * Wo[n,k] + bo[n]
// The f16 A tile (128x32 of an 8192x1024 tensor) is moved by the CDNA5
// Tensor Data Mover: one TENSOR_LOAD_TO_LDS per K-step, with the D# pad
// feature (pad 16B after every 64B row) reproducing As[128][40] exactly.
// Fallbacks: async global->LDS, then plain loads.
// ---------------------------------------------------------------------------
__global__ __launch_bounds__(256) void out_proj_kernel(
    const _Float16* __restrict__ A, const float* __restrict__ Wo,
    const float* __restrict__ bo, float* __restrict__ out)
{
  __shared__ __align__(32) _Float16 As[128][40];
  __shared__ __align__(32) _Float16 Bs[32][144];

  const int tid = threadIdx.x, lane = tid & 31, wid = tid >> 5;
  const int laneLo = lane & 15, laneHi = lane >> 4;
  const int wm = wid & 3, wn = wid >> 2;
  const int m0 = blockIdx.y * 128, n0 = blockIdx.x * 128;

#if HAVE_TDM
  const unsigned lds_As =
      (unsigned)(unsigned long long)
      (__attribute__((address_space(3))) void*)&As[0][0];
#endif

  v8f acc[2][4] = {};

  for (int k0 = 0; k0 < CE; k0 += 32) {
    __syncthreads();
    // ---- A tile ----
#if HAVE_TDM
    if (wid == 0) {
      unsigned long long ga =
          (unsigned long long)(const void*)(A + (size_t)m0 * CE + k0);
      v4u g0;
      g0[0] = 1u;                                         // count=1, user D#
      g0[1] = lds_As;                                     // LDS byte address
      g0[2] = (unsigned)(ga & 0xffffffffu);               // global_addr lo
      g0[3] = (unsigned)((ga >> 32) & 0x01ffffffu)        // global_addr hi
              | (2u << 30);                               // type=2 (image)
      v8i g1;
      g1[0] = (int)((1u << 16)      // data_size = 1 -> 2-byte elements
                  | (1u << 20)      // pad_enable
                  | (3u << 22)      // pad_interval: 16 DWORDs (one 64B row)
                  | (3u << 25));    // pad_amount: 4 DWORDs (16B -> 80B rows)
      g1[1] = (int)(((unsigned)CE & 0xffffu) << 16);      // tensor_dim0=1024
      g1[2] = (int)(((unsigned)(CB * CS) & 0xffffu) << 16); // tensor_dim1=8192
      g1[3] = (int)(32u << 16);                           // tile_dim0=32
      g1[4] = (int)128u;                                  // tile_dim1=128
      g1[5] = (int)(unsigned)CE;                          // dim0_stride=1024
      g1[6] = 0; g1[7] = 0;
      v4i gz4 = {};
      v8i gz8 = {};
      __builtin_amdgcn_tensor_load_to_lds(g0, g1, gz4, gz4, gz8, 0);
    }
#elif HAVE_ASYNC_LDS
    #pragma unroll
    for (int p = 0; p < 2; ++p) {
      int l = p * 256 + tid;                 // 0..511
      int row = l >> 2, ch = (l & 3) << 3;
      __builtin_amdgcn_global_load_async_to_lds_b128(
          (gptr_v4i)(A + (size_t)(m0 + row) * CE + k0 + ch),
          (lptr_v4i)&As[row][ch], 0, 0);
    }
#else
    #pragma unroll
    for (int p = 0; p < 2; ++p) {
      int l = p * 256 + tid;
      int row = l >> 2, ch = (l & 3) << 3;
      *(v8h*)&As[row][ch] = *(const v8h*)(A + (size_t)(m0 + row) * CE + k0 + ch);
    }
#endif
    // ---- Wo^T tile with f32->f16 conversion ----
    #pragma unroll
    for (int p = 0; p < 4; ++p) {
      int l = p * 256 + tid;
      int n = l >> 3, ch = (l & 7) << 2;
      v4f w = *(const v4f*)(Wo + (size_t)(n0 + n) * CE + k0 + ch);
      #pragma unroll
      for (int i = 0; i < 4; ++i) Bs[ch + i][n] = (_Float16)w[i];
    }
#if HAVE_TDM
    if (wid == 0) __builtin_amdgcn_s_wait_tensorcnt(0);
#elif HAVE_ASYNC_LDS
    __builtin_amdgcn_s_wait_asynccnt(0);
#endif
    __syncthreads();

    v16h a[2], b[4];
    #pragma unroll
    for (int fm = 0; fm < 2; ++fm) {
      const _Float16* pa = &As[wm * 32 + fm * 16 + laneLo][8 * laneHi];
      #pragma unroll
      for (int hh = 0; hh < 8; ++hh) { a[fm][hh] = pa[hh]; a[fm][hh + 8] = pa[hh + 16]; }
    }
    #pragma unroll
    for (int fn = 0; fn < 4; ++fn)
      for (int hh = 0; hh < 8; ++hh) { b[fn][hh] = Bs[8 * laneHi + hh][wn * 64 + fn * 16 + laneLo]; b[fn][hh + 8] = Bs[16 + 8 * laneHi + hh][wn * 64 + fn * 16 + laneLo]; }
    #pragma unroll
    for (int fm = 0; fm < 2; ++fm)
      #pragma unroll
      for (int fn = 0; fn < 4; ++fn)
        acc[fm][fn] = __builtin_amdgcn_wmma_f32_16x16x32_f16(
            false, a[fm], false, b[fn], (short)0, acc[fm][fn], false, false);
  }

  #pragma unroll
  for (int fm = 0; fm < 2; ++fm) {
    #pragma unroll
    for (int fn = 0; fn < 4; ++fn) {
      int n = n0 + wn * 64 + fn * 16 + laneLo;
      float bb = bo[n];
      #pragma unroll
      for (int r = 0; r < 8; ++r) {
        int m = m0 + wm * 32 + fm * 16 + r + 8 * laneHi;
        out[(size_t)m * CE + n] = acc[fm][fn][r] + bb;
      }
    }
  }
}

// ---------------------------------------------------------------------------
extern "C" void kernel_launch(void* const* d_in, const int* in_sizes, int n_in,
                              void* d_out, int out_size, void* d_ws, size_t ws_size,
                              hipStream_t stream) {
  (void)in_sizes; (void)n_in; (void)out_size; (void)ws_size;

  const float* query = (const float*)d_in[0];
  const float* key   = (const float*)d_in[1];
  const float* value = (const float*)d_in[2];
  const int*   mask  = (const int*)d_in[3];
  const float* Wq = (const float*)d_in[4];
  const float* bq = (const float*)d_in[5];
  const float* Wk = (const float*)d_in[6];
  const float* bk = (const float*)d_in[7];
  const float* Wv = (const float*)d_in[8];
  const float* bv = (const float*)d_in[9];
  const float* Wo = (const float*)d_in[10];
  const float* bo = (const float*)d_in[11];

  float* out   = (float*)d_out;                       // (B,Sq,E)
  float* attnW = out + (size_t)CB * CS * CE;          // (B,H,Sq,Sk)

  const size_t NE = (size_t)CB * CS * CE;             // 8,388,608 elems
  _Float16* q16 = (_Float16*)d_ws;                    // head-major (B,H,S,D)
  _Float16* k16 = q16 + NE;
  _Float16* v16 = k16 + NE;
  _Float16* o16 = v16 + NE;                           // (B,S,E) f16

  proj_qkv_kernel<<<dim3(CE / 128, (CB * CS) / 128, 3), 256, 0, stream>>>(
      query, key, value, Wq, Wk, Wv, bq, bk, bv, q16, k16, v16);

  attention_kernel<<<dim3(CS / 32, CH, CB), 256, 0, stream>>>(
      q16, k16, v16, mask, attnW, o16);

  out_proj_kernel<<<dim3(CE / 128, (CB * CS) / 128), 256, 0, stream>>>(
      o16, Wo, bo, out);
}
